// TweetyBERT_58523224375403
// MI455X (gfx1250) — matrix operand order repaired and checked by reading, not compiled, not run on hardware
//
#include <hip/hip_runtime.h>
#include <hip/hip_bf16.h>
#include <math.h>

// ---------------------------------------------------------------------------
// TweetyBERT forward for MI455X (gfx1250).
// All GEMMs: f16 operands (pre-converted), f32 accumulation via
// v_wmma_f32_16x16x32_f16. 32x64 tile per wave, ping-pong double-buffered
// K loop (K % 64 == 0 for every GEMM in this model).
// ---------------------------------------------------------------------------

typedef __attribute__((ext_vector_type(16))) _Float16 v16h;
typedef __attribute__((ext_vector_type(8)))  _Float16 v8h;
typedef __attribute__((ext_vector_type(8)))  float    v8f;

#define CB 4
#define CT 1000
#define CDM 512
#define CH 8
#define CDEPTH 64
#define CFFN 2048
#define CTP 1024   // padded attention K dim (multiple of 32)

__device__ __forceinline__ float gelu_exact(float x) {
    return 0.5f * x * (1.0f + erff(x * 0.70710678118654752f));
}

// ----- WMMA f16 fragment loaders (contiguous k order, same for A and B) -------------
// A 16x32: lanes 0-15 hold M=lane, K=0..15 contiguous; lanes 16-31 hold
// K=16..31.  One 32B load from a precomputed lane base pointer.
__device__ __forceinline__ v16h ld_a(const _Float16* __restrict__ p, int kb) {
    return *(const v16h*)(p + kb);
}
// B 32x16: lanes 0-15 hold N=lane, K=0..15 contiguous; lanes 16-31 K=16..31.
__device__ __forceinline__ v16h ld_b(const _Float16* __restrict__ p, int kb) {
    return *(const v16h*)(p + kb);
}

// ----- Generic batched GEMM: C = act(A[M,K] * W[N,K]^T + bias) (+ residual) --
// A, W are f16; accum f32; output f32 (optional residual) or f16.
// Batch z: offset = (z/bdiv)*s?0 + (z%bdiv)*s?1 (element counts).
// Block = 128 threads = 4 waves; each wave computes a 32x64 tile (8 WMMA/K-step).
// Requires K % 64 == 0.
__global__ __launch_bounds__(128) void wmma_gemm_k(
    const _Float16* __restrict__ A, const _Float16* __restrict__ W,
    const float* __restrict__ bias, const float* __restrict__ res,
    void* __restrict__ Cv,
    int M, int N, int K, int lda, int ldb, int ldc,
    long long sA0, long long sA1, long long sB0, long long sB1,
    long long sC0, long long sC1, int bdiv, int act, int outHalf)
{
    int z = blockIdx.z;
    int zq = z / bdiv, zr = z % bdiv;
    A += zq * sA0 + zr * sA1;
    W += zq * sB0 + zr * sB1;
    long long coff = zq * sC0 + zr * sC1;

    int tid  = threadIdx.x;
    int wave = tid >> 5;
    int lane = tid & 31;
    int hi   = lane >> 4;
    int l16  = lane & 15;

    int rowTile = blockIdx.y * 128 + wave * 32;
    int colBase = blockIdx.x * 64;

    int ar0 = min(rowTile + l16, M - 1);        // clamped A rows (stores masked)
    int ar1 = min(rowTile + 16 + l16, M - 1);
    int bn0 = min(colBase + l16,      N - 1);
    int bn1 = min(colBase + 16 + l16, N - 1);
    int bn2 = min(colBase + 32 + l16, N - 1);
    int bn3 = min(colBase + 48 + l16, N - 1);

    // per-lane base pointers (offset addressing in the loop)
    const _Float16* pa0 = A + (size_t)ar0 * lda + hi * 16;
    const _Float16* pa1 = A + (size_t)ar1 * lda + hi * 16;
    const _Float16* pb0 = W + (size_t)bn0 * ldb + hi * 16;
    const _Float16* pb1 = W + (size_t)bn1 * ldb + hi * 16;
    const _Float16* pb2 = W + (size_t)bn2 * ldb + hi * 16;
    const _Float16* pb3 = W + (size_t)bn3 * ldb + hi * 16;

    v8f acc00 = {0,0,0,0,0,0,0,0}, acc01 = {0,0,0,0,0,0,0,0};
    v8f acc02 = {0,0,0,0,0,0,0,0}, acc03 = {0,0,0,0,0,0,0,0};
    v8f acc10 = {0,0,0,0,0,0,0,0}, acc11 = {0,0,0,0,0,0,0,0};
    v8f acc12 = {0,0,0,0,0,0,0,0}, acc13 = {0,0,0,0,0,0,0,0};

#define WMMA8(A0, A1, B0, B1, B2, B3)                                          \
    acc00 = __builtin_amdgcn_wmma_f32_16x16x32_f16(false, A0, false, B0, (short)0, acc00, false, false); \
    acc01 = __builtin_amdgcn_wmma_f32_16x16x32_f16(false, A0, false, B1, (short)0, acc01, false, false); \
    acc02 = __builtin_amdgcn_wmma_f32_16x16x32_f16(false, A0, false, B2, (short)0, acc02, false, false); \
    acc03 = __builtin_amdgcn_wmma_f32_16x16x32_f16(false, A0, false, B3, (short)0, acc03, false, false); \
    acc10 = __builtin_amdgcn_wmma_f32_16x16x32_f16(false, A1, false, B0, (short)0, acc10, false, false); \
    acc11 = __builtin_amdgcn_wmma_f32_16x16x32_f16(false, A1, false, B1, (short)0, acc11, false, false); \
    acc12 = __builtin_amdgcn_wmma_f32_16x16x32_f16(false, A1, false, B2, (short)0, acc12, false, false); \
    acc13 = __builtin_amdgcn_wmma_f32_16x16x32_f16(false, A1, false, B3, (short)0, acc13, false, false)

    // ping-pong stages: stage a holds chunk kb, stage b holds chunk kb+32
    v16h A0a = ld_a(pa0, 0),  A1a = ld_a(pa1, 0);
    v16h B0a = ld_b(pb0, 0),  B1a = ld_b(pb1, 0);
    v16h B2a = ld_b(pb2, 0),  B3a = ld_b(pb3, 0);
    v16h A0b = ld_a(pa0, 32), A1b = ld_a(pa1, 32);
    v16h B0b = ld_b(pb0, 32), B1b = ld_b(pb1, 32);
    v16h B2b = ld_b(pb2, 32), B3b = ld_b(pb3, 32);

    int iters = K >> 6;
    for (int it = 0; it < iters - 1; ++it) {
        int k0 = (it << 6) + 64;
        int k1 = k0 + 32;
        __builtin_prefetch(pa0 + k0 + 64, 0, 3);
        __builtin_prefetch(pa1 + k0 + 64, 0, 3);
        WMMA8(A0a, A1a, B0a, B1a, B2a, B3a);
        A0a = ld_a(pa0, k0); A1a = ld_a(pa1, k0);
        B0a = ld_b(pb0, k0); B1a = ld_b(pb1, k0);
        B2a = ld_b(pb2, k0); B3a = ld_b(pb3, k0);
        WMMA8(A0b, A1b, B0b, B1b, B2b, B3b);
        A0b = ld_a(pa0, k1); A1b = ld_a(pa1, k1);
        B0b = ld_b(pb0, k1); B1b = ld_b(pb1, k1);
        B2b = ld_b(pb2, k1); B3b = ld_b(pb3, k1);
    }
    WMMA8(A0a, A1a, B0a, B1a, B2a, B3a);
    WMMA8(A0b, A1b, B0b, B1b, B2b, B3b);
#undef WMMA8

    v8f accArr[2][4] = { { acc00, acc01, acc02, acc03 },
                         { acc10, acc11, acc12, acc13 } };
#pragma unroll
    for (int i = 0; i < 2; ++i) {
#pragma unroll
        for (int j = 0; j < 4; ++j) {
            int c = colBase + j * 16 + l16;
            if (c >= N) continue;
            float bv = bias ? bias[c] : 0.0f;
#pragma unroll
            for (int v = 0; v < 8; ++v) {
                int r = rowTile + i * 16 + v + hi * 8;
                if (r < M) {
                    float val = accArr[i][j][v] + bv;
                    if (act == 1) val = fmaxf(val, 0.0f);
                    if (outHalf) {
                        _Float16* C = (_Float16*)Cv + coff;
                        C[(size_t)r * ldc + c] = (_Float16)val;
                    } else {
                        float* C = (float*)Cv + coff;
                        if (res) val += res[coff + (size_t)r * ldc + c];
                        C[(size_t)r * ldc + c] = val;
                    }
                }
            }
        }
    }
}

// ----- f32 -> f16 convert ---------------------------------------------------
__global__ __launch_bounds__(256) void cvt_f32_f16_k(
    const float* __restrict__ in, _Float16* __restrict__ out, int n)
{
    int i = blockIdx.x * 256 + threadIdx.x;
    if (i < n) out[i] = (_Float16)in[i];
}

// ----- conv1 (1->32, 5x5 SAME on 196x1000) + GELU + maxpool(14,1) -----------
__global__ __launch_bounds__(256) void conv1_pool_k(
    const float* __restrict__ spec, const float* __restrict__ w,
    const float* __restrict__ bias, float* __restrict__ out)
{
    int idx = blockIdx.x * 256 + threadIdx.x;
    if (idx >= CB * 32 * 14 * CT) return;
    int t  = idx % CT;
    int fp = (idx / CT) % 14;
    int oc = (idx / (14 * CT)) % 32;
    int b  = idx / (14 * CT * 32);
    const float* wp = w + oc * 25;
    const float* sp = spec + (size_t)b * 196 * CT;
    float m = -INFINITY;
    for (int f = 0; f < 14; ++f) {
        int fi = fp * 14 + f;
        float acc = bias[oc];
        for (int df = -2; df <= 2; ++df) {
            int ff = fi + df; if (ff < 0 || ff >= 196) continue;
            for (int dt = -2; dt <= 2; ++dt) {
                int tt = t + dt; if (tt < 0 || tt >= CT) continue;
                acc += sp[ff * CT + tt] * wp[(df + 2) * 5 + (dt + 2)];
            }
        }
        m = fmaxf(m, gelu_exact(acc));
    }
    out[idx] = m;  // [B,32,14,T] f32
}

// ----- conv2 (32->64, 5x5 SAME on 14x1000) + GELU + maxpool(14,1) -> f16 ----
__global__ __launch_bounds__(256) void conv2_pool_k(
    const float* __restrict__ in, const float* __restrict__ w,
    const float* __restrict__ bias, _Float16* __restrict__ x64)
{
    int idx = blockIdx.x * 256 + threadIdx.x;
    if (idx >= CB * CT * 64) return;
    int oc = idx & 63;
    int t  = (idx >> 6) % CT;
    int b  = idx / (64 * CT);
    float m = -INFINITY;
    for (int f = 0; f < 14; ++f) {
        float acc = bias[oc];
        for (int ic = 0; ic < 32; ++ic) {
            const float* ip = in + ((size_t)(b * 32 + ic) * 14) * CT;
            const float* wp = w + (oc * 32 + ic) * 25;
            for (int df = -2; df <= 2; ++df) {
                int ff = f + df; if (ff < 0 || ff >= 14) continue;
                for (int dt = -2; dt <= 2; ++dt) {
                    int tt = t + dt; if (tt < 0 || tt >= CT) continue;
                    acc += ip[ff * CT + tt] * wp[(df + 2) * 5 + (dt + 2)];
                }
            }
        }
        m = fmaxf(m, gelu_exact(acc));
    }
    x64[idx] = (_Float16)m;  // [(b*T+t), 64] f16
}

// ----- LayerNorm over D=512 per row: f32 in, f16 out ------------------------
__global__ __launch_bounds__(256) void layernorm_k(
    const float* __restrict__ x, const float* __restrict__ g,
    const float* __restrict__ b, _Float16* __restrict__ y, int D)
{
    int row = blockIdx.x;
    const float* xr = x + (size_t)row * D;
    _Float16* yr = y + (size_t)row * D;
    __shared__ float s1[256], s2[256];
    float sum = 0.0f, sq = 0.0f;
    for (int c = threadIdx.x; c < D; c += 256) {
        float v = xr[c]; sum += v; sq += v * v;
    }
    s1[threadIdx.x] = sum; s2[threadIdx.x] = sq;
    __syncthreads();
    for (int off = 128; off > 0; off >>= 1) {
        if (threadIdx.x < off) {
            s1[threadIdx.x] += s1[threadIdx.x + off];
            s2[threadIdx.x] += s2[threadIdx.x + off];
        }
        __syncthreads();
    }
    float mu = s1[0] / D;
    float var = s2[0] / D - mu * mu;
    float rstd = rsqrtf(var + 1e-5f);
    for (int c = threadIdx.x; c < D; c += 256)
        yr[c] = (_Float16)(g[c] * ((xr[c] - mu) * rstd) + b[c]);
}

// ----- softmax with music-transformer skew gather ---------------------------
// S: [bh][T][CTP] f32 Q*K^T (cols >= T untouched); Rq: [bh][T][CTP] f32 Q*Er^T
// Output P: [bh][T][CTP] f16, cols >= T exactly zero (pad for P*V GEMM).
__global__ __launch_bounds__(256) void softmax_skew_k(
    const float* __restrict__ Rq, const float* __restrict__ S,
    _Float16* __restrict__ P, float scale)
{
    int s  = blockIdx.x;
    int bh = blockIdx.y;
    const float* Srow = S + ((size_t)bh * CT + s) * CTP;
    const float* Rb   = Rq + (size_t)bh * CT * CTP;
    _Float16* Prow    = P + ((size_t)bh * CT + s) * CTP;
    __shared__ float red[256];
    float vals[4];
    float mx = -INFINITY;
#pragma unroll
    for (int i = 0; i < 4; ++i) {
        int c = threadIdx.x + i * 256;
        float v;
        if (c < CT) {
            float rel;
            if (c <= s) {
                rel = Rb[(size_t)s * CTP + (CT - 1 - s + c)];
            } else {
                int j = c - s - 1;
                rel = (j == 0) ? 0.0f : Rb[(size_t)(s + 1) * CTP + (j - 1)];
            }
            v = (Srow[c] + rel) * scale;
        } else v = -INFINITY;
        vals[i] = v;
        mx = fmaxf(mx, v);
    }
    red[threadIdx.x] = mx; __syncthreads();
    for (int off = 128; off > 0; off >>= 1) {
        if (threadIdx.x < off)
            red[threadIdx.x] = fmaxf(red[threadIdx.x], red[threadIdx.x + off]);
        __syncthreads();
    }
    mx = red[0]; __syncthreads();
    float sum = 0.0f;
#pragma unroll
    for (int i = 0; i < 4; ++i) {
        float e = (vals[i] == -INFINITY) ? 0.0f : expf(vals[i] - mx);
        vals[i] = e; sum += e;
    }
    red[threadIdx.x] = sum; __syncthreads();
    for (int off = 128; off > 0; off >>= 1) {
        if (threadIdx.x < off) red[threadIdx.x] += red[threadIdx.x + off];
        __syncthreads();
    }
    float inv = 1.0f / red[0];
#pragma unroll
    for (int i = 0; i < 4; ++i) {
        int c = threadIdx.x + i * 256;
        Prow[c] = (_Float16)(vals[i] * inv);
    }
}

// ----- V f16 [B*T, DM] -> Vt f16 [bh][64][CTP] (zero padded) ----------------
__global__ __launch_bounds__(256) void transpose_v_k(
    const _Float16* __restrict__ V, _Float16* __restrict__ Vt)
{
    int idx = blockIdx.x * 256 + threadIdx.x;
    if (idx >= 32 * 64 * CTP) return;
    int t  = idx & (CTP - 1);
    int d  = (idx >> 10) & 63;
    int bh = idx >> 16;
    int b = bh >> 3, h = bh & 7;
    _Float16 v = (_Float16)0.0f;
    if (t < CT) v = V[((size_t)(b * CT + t)) * CDM + h * 64 + d];
    Vt[idx] = v;
}

// ----- head buffer f16 [bh][T][64] -> f16 [B*T, DM] -------------------------
__global__ __launch_bounds__(256) void repack_heads_k(
    const _Float16* __restrict__ hbuf, _Float16* __restrict__ out)
{
    int idx = blockIdx.x * 256 + threadIdx.x;
    if (idx >= CB * CT * CDM) return;
    int c   = idx & (CDM - 1);
    int row = idx >> 9;
    int b = row / CT, t = row % CT;
    int h = c >> 6, d = c & 63;
    out[idx] = hbuf[((size_t)(b * 8 + h) * CT + t) * 64 + d];
}

// ----- host-side launcher helper -------------------------------------------
static void gemm(hipStream_t st, const _Float16* A, const _Float16* W,
                 const float* bias, const float* res, void* C,
                 int M, int N, int K, int lda, int ldb, int ldc,
                 long long sA0 = 0, long long sA1 = 0,
                 long long sB0 = 0, long long sB1 = 0,
                 long long sC0 = 0, long long sC1 = 0,
                 int batch = 1, int bdiv = 1, int act = 0, int outHalf = 0)
{
    dim3 grid((N + 63) / 64, (M + 127) / 128, batch);
    wmma_gemm_k<<<grid, 128, 0, st>>>(A, W, bias, res, C, M, N, K, lda, ldb, ldc,
                                      sA0, sA1, sB0, sB1, sC0, sC1, bdiv, act, outHalf);
}

static void cvt(hipStream_t st, const float* in, _Float16* out, int n)
{
    cvt_f32_f16_k<<<(n + 255) / 256, 256, 0, st>>>(in, out, n);
}

extern "C" void kernel_launch(void* const* d_in, const int* in_sizes, int n_in,
                              void* d_out, int out_size, void* d_ws, size_t ws_size,
                              hipStream_t stream)
{
    (void)in_sizes; (void)n_in; (void)out_size; (void)ws_size;
    const float* spec  = (const float*)d_in[0];
    const float* c1w   = (const float*)d_in[1];
    const float* c1b   = (const float*)d_in[2];
    const float* c2w   = (const float*)d_in[3];
    const float* c2b   = (const float*)d_in[4];
    const float* projw = (const float*)d_in[5];
    const float* projb = (const float*)d_in[6];
    const float* ln1g  = (const float*)d_in[7];
    const float* ln1b  = (const float*)d_in[8];
    const float* qw    = (const float*)d_in[9];
    const float* kw    = (const float*)d_in[10];
    const float* vw    = (const float*)d_in[11];
    const float* dwp   = (const float*)d_in[12];
    const float* qb    = (const float*)d_in[13];
    const float* kbb   = (const float*)d_in[14];
    const float* vbb   = (const float*)d_in[15];
    const float* dbb   = (const float*)d_in[16];
    const float* Er    = (const float*)d_in[17];
    const float* ln2g  = (const float*)d_in[18];
    const float* ln2b  = (const float*)d_in[19];
    const float* f1w   = (const float*)d_in[20];
    const float* f1b   = (const float*)d_in[21];
    const float* f2w   = (const float*)d_in[22];
    const float* f2b   = (const float*)d_in[23];
    const float* dpw   = (const float*)d_in[24];
    const float* dpb   = (const float*)d_in[25];

    // ---- workspace layout ----
    float* fws = (float*)d_ws;
    float* pool1 = fws;                  // 1,792,000 f32
    float* x     = pool1 + 1792000;      // 2,048,000 f32 (residual stream)
    float* Sb    = x + 2048000;          // 32,768,000 f32 (Q K^T)
    float* Rq    = Sb + 32768000;        // 32,768,000 f32 (Q Er^T)
    _Float16* hws = (_Float16*)(Rq + 32768000);
    _Float16* x64h = hws;                //   256,000 h
    _Float16* xnh  = x64h + 256000;      // 2,048,000 h
    _Float16* Qh   = xnh  + 2048000;     // 2,048,000 h
    _Float16* Kh   = Qh   + 2048000;     // 2,048,000 h
    _Float16* Vh   = Kh   + 2048000;     // 2,048,000 h
    _Float16* Vth  = Vh   + 2048000;     // 2,097,152 h
    _Float16* Ph   = Vth  + 2097152;     // 32,768,000 h
    _Float16* hbh  = Ph   + 32768000;    // 2,048,000 h
    _Float16* aoh  = hbh  + 2048000;     // 2,048,000 h
    _Float16* fhh  = aoh  + 2048000;     // 8,192,000 h
    _Float16* xh   = fhh  + 8192000;     // 2,048,000 h (x before deproj)
    // f16 weights
    _Float16* projwh = xh + 2048000;     //    32,768 h
    _Float16* qwh    = projwh + 32768;   // 1,048,576 h each
    _Float16* kwh    = qwh + 1048576;
    _Float16* vwh    = kwh + 1048576;
    _Float16* dwh    = vwh + 1048576;
    _Float16* Erh    = dwh + 1048576;    //   256,000 h
    _Float16* f1wh   = Erh + 256000;     // 4,194,304 h
    _Float16* f2wh   = f1wh + 4194304;   // 4,194,304 h
    _Float16* dpwh   = f2wh + 4194304;   //    65,536 h

    // ---- weight conversion (deterministic, every call) ----
    cvt(stream, projw, projwh, 512 * 64);
    cvt(stream, qw,  qwh, 4 * 512 * 512);
    cvt(stream, kw,  kwh, 4 * 512 * 512);
    cvt(stream, vw,  vwh, 4 * 512 * 512);
    cvt(stream, dwp, dwh, 4 * 512 * 512);
    cvt(stream, Er,  Erh, 4 * 1000 * 64);
    cvt(stream, f1w, f1wh, 4 * 2048 * 512);
    cvt(stream, f2w, f2wh, 4 * 2048 * 512);
    cvt(stream, dpw, dpwh, 128 * 512);

    // ---- conv frontend ----
    conv1_pool_k<<<(1792000 + 255) / 256, 256, 0, stream>>>(spec, c1w, c1b, pool1);
    conv2_pool_k<<<(256000 + 255) / 256, 256, 0, stream>>>(pool1, c2w, c2b, x64h);

    // projection 64 -> 512 (f32 out: residual stream)
    gemm(stream, x64h, projwh, projb, nullptr, x, 4000, 512, 64, 64, 64, 512);

    const long long sQ0 = (long long)CT * CDM, sQ1 = 64;
    const long long sS0 = 8LL * CT * CTP,      sS1 = (long long)CT * CTP;
    const long long sV0 = 8LL * 64 * CTP,      sV1 = 64LL * CTP;
    const long long sH0 = 8LL * CT * 64,       sH1 = (long long)CT * 64;

    for (int l = 0; l < 4; ++l) {
        layernorm_k<<<4000, 256, 0, stream>>>(x, ln1g + l * CDM, ln1b + l * CDM, xnh, CDM);

        // QKV (f16 out: feed attention GEMMs directly)
        gemm(stream, xnh, qwh + (size_t)l * CDM * CDM, qb + l * CDM, nullptr, Qh,
             4000, 512, 512, 512, 512, 512, 0,0,0,0,0,0, 1, 1, 0, /*outHalf=*/1);
        gemm(stream, xnh, kwh + (size_t)l * CDM * CDM, kbb + l * CDM, nullptr, Kh,
             4000, 512, 512, 512, 512, 512, 0,0,0,0,0,0, 1, 1, 0, 1);
        gemm(stream, xnh, vwh + (size_t)l * CDM * CDM, vbb + l * CDM, nullptr, Vh,
             4000, 512, 512, 512, 512, 512, 0,0,0,0,0,0, 1, 1, 0, 1);

        transpose_v_k<<<(32 * 64 * CTP + 255) / 256, 256, 0, stream>>>(Vh, Vth);

        // S = Q K^T (f32 out), batched over 32 (b,h)
        gemm(stream, Qh, Kh, nullptr, nullptr, Sb, 1000, 1000, 64, 512, 512, CTP,
             sQ0, sQ1, sQ0, sQ1, sS0, sS1, 32, 8, 0, 0);
        // R = Q Er^T (f32 out)
        gemm(stream, Qh, Erh + (size_t)l * CT * 64, nullptr, nullptr, Rq,
             1000, 1000, 64, 512, 64, CTP,
             sQ0, sQ1, 0, 0, sS0, sS1, 32, 8, 0, 0);

        softmax_skew_k<<<dim3(1000, 32), 256, 0, stream>>>(Rq, Sb, Ph, 0.125f);

        // head = P V (f16 out), K padded to 1024 with zero prob mass
        gemm(stream, Ph, Vth, nullptr, nullptr, hbh, 1000, 64, CTP, CTP, CTP, 64,
             sS0, sS1, sV0, sV1, sH0, sH1, 32, 8, 0, 1);

        repack_heads_k<<<(CB * CT * CDM + 255) / 256, 256, 0, stream>>>(hbh, aoh);

        // x = x + out @ dw^T + db (f32 out + residual)
        gemm(stream, aoh, dwh + (size_t)l * CDM * CDM, dbb + l * CDM, x, x,
             4000, 512, 512, 512, 512, 512);

        layernorm_k<<<4000, 256, 0, stream>>>(x, ln2g + l * CDM, ln2b + l * CDM, xnh, CDM);

        // ffh = relu(xn @ f1w^T + f1b) (f16 out)
        gemm(stream, xnh, f1wh + (size_t)l * CFFN * CDM, f1b + l * CFFN, nullptr, fhh,
             4000, 2048, 512, 512, 512, 2048, 0,0,0,0,0,0, 1, 1, /*act=*/1, /*outHalf=*/1);
        // x = x + ffh @ f2w^T + f2b (f32 out + residual)
        gemm(stream, fhh, f2wh + (size_t)l * CDM * CFFN, f2b + l * CDM, x, x,
             4000, 512, 2048, 2048, 2048, 512);
    }

    // final deprojection 512 -> 128 (convert x to f16 first)
    cvt(stream, x, xh, 4000 * 512);
    gemm(stream, xh, dpwh, dpb, nullptr, (float*)d_out, 4000, 128, 512, 512, 512, 128);
}
